// PatchAggregator_81286551044478
// MI455X (gfx1250) — matrix operand order repaired and checked by reading, not compiled, not run on hardware
//
#include <hip/hip_runtime.h>
#include <hip/hip_bf16.h>

// ---------------------------------------------------------------------------
// Problem constants (from reference)
// ---------------------------------------------------------------------------
#define BATCH 8
#define SLEN  4096
#define DDIM  1024
#define HDIM  512           // D/2
#define NPATCH 8            // MAX_N_LATENT
#define EOS_THR 0.7f
#define EOS_SOFT 0.5f
#define COS_THR 0.3f
#define EPS_V 1e-8f

typedef __attribute__((ext_vector_type(16))) __bf16 v16bf;
typedef __attribute__((ext_vector_type(8)))  float  v8f;
typedef __attribute__((ext_vector_type(2)))  float  v2f;

__device__ __forceinline__ __bf16 to_bf16(float f) { return (__bf16)f; }

#define WMMA_BF16(a, b, c) \
    __builtin_amdgcn_wmma_f32_16x16x32_bf16(false, (a), false, (b), (short)0, (c), false, false)

// ---------------------------------------------------------------------------
// Kernel 0: init scratch (norm/dot accumulators + bounds flags)
// ---------------------------------------------------------------------------
__global__ void init_kernel(float* __restrict__ nrm, float* __restrict__ dotv,
                            int* __restrict__ flags) {
    int idx = blockIdx.x * blockDim.x + threadIdx.x;
    if (idx < BATCH * SLEN) { nrm[idx] = 0.f; dotv[idx] = 0.f; }
    if (idx == 0) { flags[0] = SLEN; flags[1] = -1; flags[2] = 0; }
}

// ---------------------------------------------------------------------------
// Kernel 1: convert eos_w1 (f32, [k][n] 1024x512) -> bf16 in WMMA B-fragment
// order: w1sw[((c*32 + ntile)*32 + lane)*16 + i]
// B 32x16 (16-bit) layout: lanes 0-15: k=i (i<8), 16+i-8 (i>=8), n=lane; lanes 16-31: k=8+i (i<8), 24+i-8 (i>=8)
// ---------------------------------------------------------------------------
__global__ void convert_w1_kernel(const float* __restrict__ w1,
                                  __bf16* __restrict__ w1sw) {
    int idx = blockIdx.x * blockDim.x + threadIdx.x;   // 0 .. 524287
    int i    = idx & 15;
    int lane = (idx >> 4) & 31;
    int t    = (idx >> 9) & 31;
    int c    = idx >> 14;
    int v = i >> 1, h = i & 1;
    int kk = (i & 7) + ((i & 8) ? 16 : 0) + ((lane & 16) ? 8 : 0);
    int k = c * 32 + kk;
    int n = t * 16 + (lane & 15);
    w1sw[idx] = to_bf16(w1[k * HDIM + n]);
}

// ---------------------------------------------------------------------------
// Kernel 2: per-batch exclusive cumsum of mask (seg ids) + bucket counts
// ---------------------------------------------------------------------------
__global__ __launch_bounds__(256) void seg_scan_kernel(
    const int* __restrict__ mask, int* __restrict__ seg,
    float* __restrict__ counts) {
    __shared__ int tsum[256];
    __shared__ int cnt[NPATCH];
    int b = blockIdx.x, tid = threadIdx.x;
    const int* mrow = mask + b * SLEN;
    int s0 = tid * 16;
    int local[16];
    int acc = 0;
    #pragma unroll
    for (int e = 0; e < 16; ++e) { local[e] = acc; acc += mrow[s0 + e]; }
    tsum[tid] = acc;
    if (tid < NPATCH) cnt[tid] = 0;
    __syncthreads();
    for (int off = 1; off < 256; off <<= 1) {
        int v = tsum[tid];
        int u = (tid >= off) ? tsum[tid - off] : 0;
        __syncthreads();
        tsum[tid] = v + u;
        __syncthreads();
    }
    int prefix = (tid == 0) ? 0 : tsum[tid - 1];
    #pragma unroll
    for (int e = 0; e < 16; ++e) {
        int sg = prefix + local[e];
        seg[b * SLEN + s0 + e] = sg;
        if (sg < NPATCH) atomicAdd(&cnt[sg], 1);
    }
    __syncthreads();
    if (tid < NPATCH) counts[b * NPATCH + tid] = (float)cnt[tid];
}

// ---------------------------------------------------------------------------
// Kernel 3: fused EOS MLP.  Each block: 32 rows of X (two 16-row M tiles).
// 8 waves; wave w owns N columns [w*64, w*64+64) of H=512 as 4 n-tiles,
// for BOTH m-tiles (B fragments reused -> halves W1 L2 traffic).
// Main loop is manually double-buffered (ping-pong) so chunk c+1's A (LDS)
// and B (global/L2) fragment loads are in flight while chunk c's 8 WMMAs run.
// ---------------------------------------------------------------------------
__global__ __launch_bounds__(256) void eos_mlp_kernel(
    const float* __restrict__ latent, const __bf16* __restrict__ w1sw,
    const float* __restrict__ b1, const float* __restrict__ w2,
    const float* __restrict__ b2, float* __restrict__ scores) {
    __shared__ __align__(32) __bf16 ldsA[2 * 32 * 32 * 16]; // [mtile][c][lane][i], 64 KB
    __shared__ float hsum[32];
    int tid = threadIdx.x;
    int g0 = blockIdx.x * 32;               // first flattened (b,s) row
    if (tid < 32) hsum[tid] = 0.f;

    // --- stage A tile: coalesced f32 reads, packed b128 LDS writes ----------
    {
        int r  = tid >> 3;                   // row 0..31
        int c0 = (tid & 7) * 128;            // 128 k's per thread
        int mhalf = r >> 4;                  // which 16-row m-tile
        int rr = r & 15;
        const float* rowp = latent + (size_t)(g0 + r) * DDIM;
        #pragma unroll 4
        for (int e = 0; e < 128; e += 8) {
            int k = c0 + e;                  // multiple of 8
            float4 va = *(const float4*)(rowp + k);
            float4 vb = *(const float4*)(rowp + k + 4);
            int c  = k >> 5;
            int kk = k & 31;                 // multiple of 8
            int hi = (kk >> 3) & 1;          // which 16-lane half
            int base = kk - (hi ? 8 : 0);    // 0 or 16
            int i  = (base < 8) ? base : (base - 8);  // 0 or 8
            int lane = rr + (hi ? 16 : 0);
            union { __bf16 h[8]; uint4 u; } pk;
            pk.h[0] = to_bf16(va.x); pk.h[1] = to_bf16(va.y);
            pk.h[2] = to_bf16(va.z); pk.h[3] = to_bf16(va.w);
            pk.h[4] = to_bf16(vb.x); pk.h[5] = to_bf16(vb.y);
            pk.h[6] = to_bf16(vb.z); pk.h[7] = to_bf16(vb.w);
            *(uint4*)&ldsA[mhalf * 16384 + ((c * 32) + lane) * 16 + i] = pk.u;
        }
    }
    __syncthreads();

    int wave = tid >> 5, lane = tid & 31;
    int lm = lane & 15, lh = (lane >> 4) & 1;

    const __bf16* ap0 = &ldsA[lane * 16];            // m-tile 0, chunk stride 512 el
    const __bf16* ap1 = ap0 + 16384;                 // m-tile 1
    const __bf16* wp  = w1sw + ((size_t)(wave * 4) * 32 + lane) * 16; // chunk stride 16384 el

    v8f acc[2][4] = {};

    // prologue: chunk 0 into buffer A
    v16bf aA0 = *(const v16bf*)ap0;
    v16bf aA1 = *(const v16bf*)ap1;
    v16bf bA0 = *(const v16bf*)(wp);
    v16bf bA1 = *(const v16bf*)(wp + 512);
    v16bf bA2 = *(const v16bf*)(wp + 1024);
    v16bf bA3 = *(const v16bf*)(wp + 1536);

    #pragma unroll 1
    for (int c = 0; c < 32; c += 2) {
        // issue chunk c+1 loads (buffer B)
        v16bf aB0 = *(const v16bf*)(ap0 + 512);
        v16bf aB1 = *(const v16bf*)(ap1 + 512);
        v16bf bB0 = *(const v16bf*)(wp + 16384);
        v16bf bB1 = *(const v16bf*)(wp + 16384 + 512);
        v16bf bB2 = *(const v16bf*)(wp + 16384 + 1024);
        v16bf bB3 = *(const v16bf*)(wp + 16384 + 1536);
        // compute chunk c (buffer A)
        acc[0][0] = WMMA_BF16(aA0, bA0, acc[0][0]);
        acc[0][1] = WMMA_BF16(aA0, bA1, acc[0][1]);
        acc[0][2] = WMMA_BF16(aA0, bA2, acc[0][2]);
        acc[0][3] = WMMA_BF16(aA0, bA3, acc[0][3]);
        acc[1][0] = WMMA_BF16(aA1, bA0, acc[1][0]);
        acc[1][1] = WMMA_BF16(aA1, bA1, acc[1][1]);
        acc[1][2] = WMMA_BF16(aA1, bA2, acc[1][2]);
        acc[1][3] = WMMA_BF16(aA1, bA3, acc[1][3]);
        ap0 += 1024; ap1 += 1024; wp += 32768;
        // issue chunk c+2 loads (buffer A) -- last iter over-reads harmlessly
        aA0 = *(const v16bf*)ap0;
        aA1 = *(const v16bf*)ap1;
        bA0 = *(const v16bf*)(wp);
        bA1 = *(const v16bf*)(wp + 512);
        bA2 = *(const v16bf*)(wp + 1024);
        bA3 = *(const v16bf*)(wp + 1536);
        // compute chunk c+1 (buffer B)
        acc[0][0] = WMMA_BF16(aB0, bB0, acc[0][0]);
        acc[0][1] = WMMA_BF16(aB0, bB1, acc[0][1]);
        acc[0][2] = WMMA_BF16(aB0, bB2, acc[0][2]);
        acc[0][3] = WMMA_BF16(aB0, bB3, acc[0][3]);
        acc[1][0] = WMMA_BF16(aB1, bB0, acc[1][0]);
        acc[1][1] = WMMA_BF16(aB1, bB1, acc[1][1]);
        acc[1][2] = WMMA_BF16(aB1, bB2, acc[1][2]);
        acc[1][3] = WMMA_BF16(aB1, bB3, acc[1][3]);
    }

    // --- epilogue: relu + dot with w2, reduce over N ------------------------
    #pragma unroll
    for (int m = 0; m < 2; ++m) {
        float rowpart[8];
        #pragma unroll
        for (int j = 0; j < 8; ++j) rowpart[j] = 0.f;
        #pragma unroll
        for (int t = 0; t < 4; ++t) {
            int n = (wave * 4 + t) * 16 + lm;
            float bb = b1[n], ww = w2[n];
            #pragma unroll
            for (int j = 0; j < 8; ++j) {
                float hv = acc[m][t][j] + bb;
                hv = hv > 0.f ? hv : 0.f;
                rowpart[j] += hv * ww;
            }
        }
        #pragma unroll
        for (int j = 0; j < 8; ++j) {
            float v = rowpart[j];
            #pragma unroll
            for (int off = 8; off >= 1; off >>= 1) v += __shfl_xor(v, off, 16);
            if (lm == 0) atomicAdd(&hsum[m * 16 + j + lh * 8], v);
        }
    }
    __syncthreads();
    if (tid < 32) {
        float logit = hsum[tid] + b2[0];
        scores[g0 + tid] = 1.f / (1.f + __expf(-logit));
    }
}

// ---------------------------------------------------------------------------
// Kernel 4: fused segment pooling + per-row norm^2 + adjacent-row dot.
// One pass over the 128MB latent tensor. Block = (batch, 256-d slab).
// ---------------------------------------------------------------------------
__global__ __launch_bounds__(256) void pool_dot_kernel(
    const float* __restrict__ latent, const int* __restrict__ seg,
    const float* __restrict__ counts, float* __restrict__ pooled,
    float* __restrict__ nrm, float* __restrict__ dotv) {
    int b = blockIdx.x >> 2;
    int d = (blockIdx.x & 3) * 256 + threadIdx.x;
    int lane = threadIdx.x & 31;
    const float* base = latent + (size_t)b * SLEN * DDIM + d;
    const int* segb = seg + b * SLEN;
    float acc[NPATCH];
    #pragma unroll
    for (int p = 0; p < NPATCH; ++p) acc[p] = 0.f;
    float xprev = 0.f;
    #pragma unroll 1
    for (int s = 0; s < SLEN; ++s) {
        if (s + 16 < SLEN)
            __builtin_prefetch(base + (size_t)(s + 16) * DDIM, 0, 1);
        float x = base[(size_t)s * DDIM];
        int sg = segb[s];
        #pragma unroll
        for (int p = 0; p < NPATCH; ++p) acc[p] += (sg == p) ? x : 0.f;
        float nn = x * x;
        float dd = x * xprev;
        #pragma unroll
        for (int off = 16; off >= 1; off >>= 1) {
            nn += __shfl_xor(nn, off, 32);
            dd += __shfl_xor(dd, off, 32);
        }
        if (lane == 0) {
            atomicAdd(&nrm[b * SLEN + s], nn);
            if (s > 0) atomicAdd(&dotv[b * SLEN + s], dd);
        }
        xprev = x;
    }
    #pragma unroll
    for (int p = 0; p < NPATCH; ++p) {
        float cv = counts[b * NPATCH + p];
        pooled[(size_t)(b * NPATCH + p) * DDIM + d] = acc[p] / fmaxf(cv, 1.f);
    }
}

// ---------------------------------------------------------------------------
// Kernel 5: EOS decision (run-of-3 + cosine rule) -> global first/last/has
// ---------------------------------------------------------------------------
__global__ void decide_kernel(const float* __restrict__ scores,
                              const float* __restrict__ nrm,
                              const float* __restrict__ dotv,
                              int* __restrict__ flags) {
    int idx = blockIdx.x * blockDim.x + threadIdx.x;  // flattened (b,s)
    int s = idx & (SLEN - 1);
    float sc = scores[idx];
    bool any = false;
    if (s >= 2) {
        bool run = (sc > EOS_THR) && (scores[idx - 1] > EOS_THR) &&
                   (scores[idx - 2] > EOS_THR);
        float na = fmaxf(sqrtf(nrm[idx - 1]), EPS_V);
        float nb = fmaxf(sqrtf(nrm[idx]), EPS_V);
        float cosv = dotv[idx] / (na * nb);
        bool crule = (cosv < COS_THR) && (sc > EOS_SOFT);
        any = run || crule;
    }
    if (any) {
        atomicMin(&flags[0], s);
        atomicMax(&flags[1], s);
        atomicExch(&flags[2], 1);
    }
}

__global__ void finalize_kernel(const int* __restrict__ flags,
                                float* __restrict__ out_bounds) {
    if (flags[2]) {
        int start = max(0, flags[0] - 2);
        int end   = min(SLEN - 1, flags[1] + 2);
        out_bounds[0] = (float)start;
        out_bounds[1] = (float)end;
    } else {
        out_bounds[0] = -1.f;
        out_bounds[1] = -1.f;
    }
}

// ---------------------------------------------------------------------------
// Kernel 6: patches = pooled(64x1024) @ proj_w(1024x1024) + proj_b
// Full f32 precision via V_WMMA_F32_16X16X4_F32.
// Grid: 32 blocks = 4 m-tiles x 8 n-groups; each wave owns one 16-wide
// n-tile and the full K, with a 16x unrolled K loop so ~32 B-loads are in
// flight per wait batch (kills the serial L2 latency chain of round 1).
// ---------------------------------------------------------------------------
__global__ __launch_bounds__(256) void patches_kernel(
    const float* __restrict__ pooled, const float* __restrict__ projw,
    const float* __restrict__ projb, float* __restrict__ out) {
    __shared__ float ldsP[16 * DDIM];   // 64 KB
    int tid = threadIdx.x;
    int mtile = blockIdx.x >> 3;        // 0..3
    int ng    = blockIdx.x & 7;         // 0..7
    const float* src = pooled + (size_t)mtile * 16 * DDIM;
    for (int e = tid * 4; e < 16 * DDIM; e += 256 * 4)
        *(float4*)&ldsP[e] = *(const float4*)(src + e);
    __syncthreads();

    int wave = tid >> 5, lane = tid & 31;
    int lm = lane & 15, lh = (lane >> 4) & 1;
    int nt = ng * 8 + wave;             // 0..63
    int n = nt * 16 + lm;
    v8f acc = {};
#if __has_builtin(__builtin_amdgcn_wmma_f32_16x16x4_f32)
    const float* ap = &ldsP[lm * DDIM + lh * 2];
    const float* bp = projw + (size_t)(lh * 2) * DDIM + n;
    #pragma unroll 16
    for (int k0 = 0; k0 < DDIM; k0 += 4) {
        // A 16x4 f32: lanes 0-15 -> K0,K1; lanes 16-31 -> K2,K3
        v2f a; a.x = ap[k0]; a.y = ap[k0 + 1];
        // B 4x16 f32 (mirrored layout), N = lane&15
        v2f bv;
        bv.x = bp[(size_t)k0 * DDIM];
        bv.y = bp[(size_t)(k0 + 1) * DDIM];
        acc = __builtin_amdgcn_wmma_f32_16x16x4_f32(
            false, a, false, bv, (short)0, acc, false, false);
    }
    float bb = projb[n];
    #pragma unroll
    for (int j = 0; j < 8; ++j) {
        int m = j + lh * 8;
        out[(size_t)(mtile * 16 + m) * DDIM + n] = acc[j] + bb;
    }
#else
    float accs[8];
    #pragma unroll
    for (int j = 0; j < 8; ++j) accs[j] = 0.f;
    for (int k = 0; k < DDIM; ++k) {
        float w = projw[(size_t)k * DDIM + n];
        #pragma unroll
        for (int j = 0; j < 8; ++j)
            accs[j] += ldsP[(j + lh * 8) * DDIM + k] * w;
    }
    float bb = projb[n];
    #pragma unroll
    for (int j = 0; j < 8; ++j)
        out[(size_t)(mtile * 16 + j + lh * 8) * DDIM + n] = accs[j] + bb;
#endif
}

// ---------------------------------------------------------------------------
// Host-side launcher
// ---------------------------------------------------------------------------
extern "C" void kernel_launch(void* const* d_in, const int* in_sizes, int n_in,
                              void* d_out, int out_size, void* d_ws, size_t ws_size,
                              hipStream_t stream) {
    const float* latent = (const float*)d_in[0];
    const int*   mask   = (const int*)d_in[1];
    const float* projw  = (const float*)d_in[2];
    const float* projb  = (const float*)d_in[3];
    const float* w1     = (const float*)d_in[4];
    const float* b1     = (const float*)d_in[5];
    const float* w2     = (const float*)d_in[6];
    const float* b2     = (const float*)d_in[7];

    char* ws = (char*)d_ws;
    size_t off = 0;
    __bf16* w1sw   = (__bf16*)(ws + off); off += (size_t)DDIM * HDIM * 2;      // 1 MB
    float*  scores = (float*)(ws + off);  off += (size_t)BATCH * SLEN * 4;     // 128 KB
    int*    seg    = (int*)(ws + off);    off += (size_t)BATCH * SLEN * 4;     // 128 KB
    float*  counts = (float*)(ws + off);  off += 256;
    float*  nrm    = (float*)(ws + off);  off += (size_t)BATCH * SLEN * 4;     // 128 KB
    float*  dotv   = (float*)(ws + off);  off += (size_t)BATCH * SLEN * 4;     // 128 KB
    float*  pooled = (float*)(ws + off);  off += (size_t)BATCH * NPATCH * DDIM * 4; // 256 KB
    int*    flags  = (int*)(ws + off);    off += 256;

    float* patches_out = (float*)d_out;
    float* bounds_out  = (float*)d_out + (size_t)BATCH * NPATCH * DDIM;

    init_kernel<<<(BATCH * SLEN + 255) / 256, 256, 0, stream>>>(nrm, dotv, flags);
    convert_w1_kernel<<<(DDIM * HDIM) / 256, 256, 0, stream>>>(w1, w1sw);
    seg_scan_kernel<<<BATCH, 256, 0, stream>>>(mask, seg, counts);
    eos_mlp_kernel<<<(BATCH * SLEN) / 32, 256, 0, stream>>>(latent, w1sw, b1, w2, b2, scores);
    pool_dot_kernel<<<BATCH * 4, 256, 0, stream>>>(latent, seg, counts, pooled, nrm, dotv);
    decide_kernel<<<(BATCH * SLEN) / 256, 256, 0, stream>>>(scores, nrm, dotv, flags);
    patches_kernel<<<32, 256, 0, stream>>>(pooled, projw, projb, patches_out);
    finalize_kernel<<<1, 1, 0, stream>>>(flags, bounds_out);
}
